// GPTrans_3384434230000
// MI455X (gfx1250) — compile-verified, matrix operand order checked by reading, not run on hardware
//
#include <hip/hip_runtime.h>

#define DEVINL __device__ __forceinline__

typedef __bf16 bf16;
typedef __attribute__((ext_vector_type(16))) __bf16 v16bf;
typedef __attribute__((ext_vector_type(8)))  float  v8f;

constexpr int BB = 2, C = 320, N = 1024, NH = 8, DH = 40, DHP = 64, L = 2;
constexpr int AA = 16, EPD = 64;
constexpr int TWOC = 2 * C, THREEC = 3 * C;
constexpr float EDGE_SCALE = 0.17677669529663687f;  // 32^-0.5
constexpr float ATT_SCALE  = 0.15811388300841897f;  // 40^-0.5
constexpr float EPSLN = 1e-5f;

// ---------------- reductions (wave32) ----------------
DEVINL float block_reduce_max(float v, float* sm) {
#pragma unroll
  for (int o = 16; o > 0; o >>= 1) v = fmaxf(v, __shfl_xor(v, o));
  int w = threadIdx.x >> 5, lid = threadIdx.x & 31, nw = blockDim.x >> 5;
  if (lid == 0) sm[w] = v;
  __syncthreads();
  float r = (lid < nw) ? sm[lid] : -3.0e38f;
#pragma unroll
  for (int o = 16; o > 0; o >>= 1) r = fmaxf(r, __shfl_xor(r, o));
  __syncthreads();
  return r;
}
DEVINL float block_reduce_sum(float v, float* sm) {
#pragma unroll
  for (int o = 16; o > 0; o >>= 1) v += __shfl_xor(v, o);
  int w = threadIdx.x >> 5, lid = threadIdx.x & 31, nw = blockDim.x >> 5;
  if (lid == 0) sm[w] = v;
  __syncthreads();
  float r = (lid < nw) ? sm[lid] : 0.f;
#pragma unroll
  for (int o = 16; o > 0; o >>= 1) r += __shfl_xor(r, o);
  __syncthreads();
  return r;
}

// ---------------- WMMA helpers ----------------
// 16-bit A-matrix 16x32 lane layout (ISA 7.12.2); same pattern serves the
// B operand when B is supplied as W[Nout,K] row-major (D = A @ W^T).
// Per lane this is two contiguous 16B chunks -> two global_load_b128.
DEVINL v16bf load_tile(const bf16* __restrict__ p, int ld) {
  int lane = threadIdx.x & 31;
  const bf16* row = p + (lane & 15) * ld + ((lane >> 4) << 3);
  v16bf out;
#pragma unroll
  for (int e = 0; e < 16; ++e) out[e] = row[e + ((e >= 8) ? 8 : 0)];
  return out;
}

DEVINL void zero_acc(v8f acc[4]) {
  v8f z = {0.f, 0.f, 0.f, 0.f, 0.f, 0.f, 0.f, 0.f};
#pragma unroll
  for (int t = 0; t < 4; ++t) acc[t] = z;
}

DEVINL v8f wmma_bf16(v16bf a, v16bf b, v8f c) {
  return __builtin_amdgcn_wmma_f32_16x16x32_bf16(false, a, false, b, (short)0, c,
                                                 false, false);
}

// One wave computes a 16x64 strip of D = A[16,K] @ W[64,K]^T.
// Software-pipelined with the last K-step peeled: each next-step fragment
// load is issued between WMMAs so loads overlap math (partial waits) and
// each b_i dies exactly where its replacement is defined (no rotation movs).
DEVINL void gemm_strip(const bf16* __restrict__ A, int lda,
                       const bf16* __restrict__ W, int ldw,
                       int ksteps, v8f acc[4]) {
  v16bf a  = load_tile(A, lda);
  v16bf b0 = load_tile(W + 0 * 16 * ldw, ldw);
  v16bf b1 = load_tile(W + 1 * 16 * ldw, ldw);
  v16bf b2 = load_tile(W + 2 * 16 * ldw, ldw);
  v16bf b3 = load_tile(W + 3 * 16 * ldw, ldw);
  for (int ks = 0; ks < ksteps - 1; ++ks) {
    const bf16* A2 = A + (ks + 1) * 32;
    const bf16* W2 = W + (ks + 1) * 32;
    v16bf an = load_tile(A2, lda);
    acc[0] = wmma_bf16(a, b0, acc[0]);
    v16bf n0 = load_tile(W2 + 0 * 16 * ldw, ldw);
    acc[1] = wmma_bf16(a, b1, acc[1]);
    v16bf n1 = load_tile(W2 + 1 * 16 * ldw, ldw);
    acc[2] = wmma_bf16(a, b2, acc[2]);
    v16bf n2 = load_tile(W2 + 2 * 16 * ldw, ldw);
    acc[3] = wmma_bf16(a, b3, acc[3]);
    v16bf n3 = load_tile(W2 + 3 * 16 * ldw, ldw);
    a = an; b0 = n0; b1 = n1; b2 = n2; b3 = n3;
  }
  acc[0] = wmma_bf16(a, b0, acc[0]);
  acc[1] = wmma_bf16(a, b1, acc[1]);
  acc[2] = wmma_bf16(a, b2, acc[2]);
  acc[3] = wmma_bf16(a, b3, acc[3]);
}

// ---------------- small utility kernels ----------------
__global__ void k_cvt_bf16(const float* __restrict__ src, bf16* __restrict__ dst, int n) {
  int i = blockIdx.x * blockDim.x + threadIdx.x;
  if (i < n) dst[i] = (bf16)src[i];
}

__global__ void k_zero_bf16(bf16* __restrict__ dst, int n) {
  int i = blockIdx.x * blockDim.x + threadIdx.x;
  if (i < n) dst[i] = (bf16)0.f;
}

__global__ void k_transpose_in(const float* __restrict__ x, float* __restrict__ xt,
                               bf16* __restrict__ xt_bf) {
  int i = blockIdx.x * blockDim.x + threadIdx.x;
  if (i >= BB * N * C) return;
  int c = i % C, n = (i / C) % N, b = i / (C * N);
  float v = x[((size_t)b * C + c) * N + n];
  xt[i] = v;
  xt_bf[i] = (bf16)v;
}

__global__ void k_da_s(const float* __restrict__ dap, const float* __restrict__ fw,
                       const float* __restrict__ fb, float* __restrict__ s) {
  int i = blockIdx.x * blockDim.x + threadIdx.x;
  if (i >= BB * C) return;
  int c = i % C, b = i / C;
  float acc = (float)AA * fb[c];
  for (int a = 0; a < AA; ++a)
    for (int e = 0; e < EPD; ++e)
      acc += dap[(b * AA + a) * EPD + e] * fw[c * EPD + e];
  s[i] = acc;
}

__global__ void k_final(const float* __restrict__ xt, float* __restrict__ out) {
  int i = blockIdx.x * blockDim.x + threadIdx.x;
  if (i >= BB * C * N) return;
  int n = i % N, c = (i / N) % C, b = i / (C * N);
  out[i] = xt[((size_t)b * N + n) * C + c];
}

// ---------------- GEMM kernels (block = 1 wave = 32) ----------------
__global__ void __launch_bounds__(32, 2)
k_gemm_qk(const bf16* __restrict__ xt_bf, const bf16* __restrict__ wq,
          bf16* __restrict__ qk_bf) {
  int tm = blockIdx.x * 16, tn = blockIdx.y * 64, b = blockIdx.z;
  v8f acc[4]; zero_acc(acc);
  gemm_strip(xt_bf + ((size_t)b * N + tm) * C, C, wq + (size_t)tn * C, C, C / 32, acc);
  int lane = threadIdx.x & 31, cn = lane & 15, mh = (lane >> 4) * 8;
#pragma unroll
  for (int t = 0; t < 4; ++t)
#pragma unroll
    for (int r = 0; r < 8; ++r) {
      int row = tm + mh + r, col = tn + t * 16 + cn;
      qk_bf[((size_t)b * N + row) * TWOC + col] = (bf16)acc[t][r];
    }
}

__global__ void __launch_bounds__(32, 2)
k_gemm_edge(const bf16* __restrict__ qk_bf, float* __restrict__ edge) {
  int tm = blockIdx.x * 16, tn = blockIdx.y * 64, b = blockIdx.z;
  const bf16* base = qk_bf + (size_t)b * N * TWOC;
  v8f acc[4]; zero_acc(acc);
  gemm_strip(base + (size_t)tm * TWOC, TWOC, base + C + (size_t)tn * TWOC, TWOC, C / 32, acc);
  int lane = threadIdx.x & 31, cn = lane & 15, mh = (lane >> 4) * 8;
#pragma unroll
  for (int t = 0; t < 4; ++t)
#pragma unroll
    for (int r = 0; r < 8; ++r) {
      int row = tm + mh + r, col = tn + t * 16 + cn;
      edge[((size_t)b * N + row) * N + col] = acc[t][r] * EDGE_SCALE;
    }
}

__global__ void k_softmax_rows(float* __restrict__ rows) {  // grid BB*N, block 256
  __shared__ float sm[16];
  float* p = rows + (size_t)blockIdx.x * N;
  int t = threadIdx.x;
  float xv[4];
#pragma unroll
  for (int j = 0; j < 4; ++j) xv[j] = p[t + j * 256];
  float mx = fmaxf(fmaxf(xv[0], xv[1]), fmaxf(xv[2], xv[3]));
  mx = block_reduce_max(mx, sm);
  float ex[4], se = 0.f;
#pragma unroll
  for (int j = 0; j < 4; ++j) { ex[j] = __expf(xv[j] - mx); se += ex[j]; }
  se = block_reduce_sum(se, sm);
  float inv = 1.f / se;
#pragma unroll
  for (int j = 0; j < 4; ++j) p[t + j * 256] = ex[j] * inv;
}

// fused LN1 -> node2 (edge diag * s) -> gLN -> y (bf16)   block = C = 320 threads
__global__ void k_norm_node(const float* __restrict__ xt, const float* __restrict__ edge,
                            const float* __restrict__ s,
                            const float* __restrict__ ln1_g, const float* __restrict__ ln1_b,
                            const float* __restrict__ gln_g, const float* __restrict__ gln_b,
                            bf16* __restrict__ y_bf, int l) {
  __shared__ float sm[16];
  int bn = blockIdx.x, b = bn / N, n = bn % N, c = threadIdx.x;
  float xv = xt[(size_t)bn * C + c];
  float mean = block_reduce_sum(xv, sm) * (1.f / C);
  float d0 = xv - mean;
  float var = block_reduce_sum(d0 * d0, sm) * (1.f / C);
  float nt = d0 * rsqrtf(var + EPSLN) * ln1_g[l * C + c] + ln1_b[l * C + c];
  float diag = edge[((size_t)b * N + n) * N + n];
  float node2 = diag * nt * s[b * C + c] + nt;
  float mean2 = block_reduce_sum(node2, sm) * (1.f / C);
  float d2 = node2 - mean2;
  float var2 = block_reduce_sum(d2 * d2, sm) * (1.f / C);
  float y = d2 * rsqrtf(var2 + EPSLN) * gln_g[l * C + c] + gln_b[l * C + c];
  y_bf[(size_t)bn * C + c] = (bf16)y;
}

// qkv GEMM; epilogue scatters into q[B,H,N,DHP], k[B,H,N,DHP], vT[B,H,DHP,N]
__global__ void __launch_bounds__(32, 2)
k_gemm_qkv(const bf16* __restrict__ y_bf, const bf16* __restrict__ wqkv,
           const float* __restrict__ bqkv,
           bf16* __restrict__ q_bf, bf16* __restrict__ k_bf,
           bf16* __restrict__ vT_bf, int l) {
  int tm = blockIdx.x * 16, tn = blockIdx.y * 64, b = blockIdx.z;
  v8f acc[4]; zero_acc(acc);
  gemm_strip(y_bf + ((size_t)b * N + tm) * C, C,
             wqkv + ((size_t)l * THREEC + tn) * C, C, C / 32, acc);
  int lane = threadIdx.x & 31, cn = lane & 15, mh = (lane >> 4) * 8;
#pragma unroll
  for (int t = 0; t < 4; ++t)
#pragma unroll
    for (int r = 0; r < 8; ++r) {
      int row = tm + mh + r, col = tn + t * 16 + cn;
      float v = acc[t][r] + bqkv[l * THREEC + col];
      int which = col / C, j = col % C, h = j / DH, d = j % DH;
      size_t bh = (size_t)(b * NH + h);
      if (which == 0)      q_bf[(bh * N + row) * DHP + d] = (bf16)v;
      else if (which == 1) k_bf[(bh * N + row) * DHP + d] = (bf16)v;
      else                 vT_bf[(bh * DHP + d) * N + row] = (bf16)v;
    }
}

// attn logits: q@k^T * SCALE + edge*exp_w + exp_b
__global__ void __launch_bounds__(32, 2)
k_gemm_attn(const bf16* __restrict__ q_bf, const bf16* __restrict__ k_bf,
            const float* __restrict__ edge, const float* __restrict__ expw,
            const float* __restrict__ expb, float* __restrict__ attn, int l) {
  int tm = blockIdx.x * 16, tn = blockIdx.y * 64;
  int bh = blockIdx.z, b = bh / NH, h = bh % NH;
  v8f acc[4]; zero_acc(acc);
  gemm_strip(q_bf + ((size_t)bh * N + tm) * DHP, DHP,
             k_bf + ((size_t)bh * N + tn) * DHP, DHP, DHP / 32, acc);
  float ew = expw[l * NH + h], eb = expb[l * NH + h];
  int lane = threadIdx.x & 31, cn = lane & 15, mh = (lane >> 4) * 8;
#pragma unroll
  for (int t = 0; t < 4; ++t)
#pragma unroll
    for (int r = 0; r < 8; ++r) {
      int row = tm + mh + r, col = tn + t * 16 + cn;
      float logit = acc[t][r] * ATT_SCALE + edge[((size_t)b * N + row) * N + col] * ew + eb;
      attn[((size_t)bh * N + row) * N + col] = logit;
    }
}

// fused: per (b,n) over all heads: softmax -> a_bf; edge_new = sum_h (a+attn)*red_w + red_b;
// ws-softmax of edge_new -> wsum; edge += edge_new.       block = 256
__global__ void k_softmax_edge_fused(const float* __restrict__ attn, bf16* __restrict__ a_bf,
                                     float* __restrict__ edge, float* __restrict__ wsum,
                                     const float* __restrict__ redw,
                                     const float* __restrict__ redb, int l) {
  __shared__ float sm[16];
  int bn = blockIdx.x, b = bn / N, n = bn % N, t = threadIdx.x;
  float eacc[4] = {0.f, 0.f, 0.f, 0.f};
  for (int h = 0; h < NH; ++h) {
    size_t roff = (((size_t)(b * NH + h)) * N + n) * (size_t)N;
    const float* row = attn + roff;
    float xv[4];
#pragma unroll
    for (int j = 0; j < 4; ++j) xv[j] = row[t + j * 256];
    float mx = fmaxf(fmaxf(xv[0], xv[1]), fmaxf(xv[2], xv[3]));
    mx = block_reduce_max(mx, sm);
    float ex[4], se = 0.f;
#pragma unroll
    for (int j = 0; j < 4; ++j) { ex[j] = __expf(xv[j] - mx); se += ex[j]; }
    se = block_reduce_sum(se, sm);
    float inv = 1.f / se, rw = redw[l * NH + h];
    bf16* arow = a_bf + roff;
#pragma unroll
    for (int j = 0; j < 4; ++j) {
      float a = ex[j] * inv;
      arow[t + j * 256] = (bf16)a;
      eacc[j] += (a + xv[j]) * rw;
    }
  }
  float rb = redb[l], en[4], mx2 = -3.0e38f;
#pragma unroll
  for (int j = 0; j < 4; ++j) { en[j] = eacc[j] + rb; mx2 = fmaxf(mx2, en[j]); }
  mx2 = block_reduce_max(mx2, sm);
  float t1 = 0.f, t2 = 0.f;
#pragma unroll
  for (int j = 0; j < 4; ++j) {
    float e = __expf(en[j] - mx2);
    t1 += e; t2 += e * en[j];
  }
  t1 = block_reduce_sum(t1, sm);
  t2 = block_reduce_sum(t2, sm);
  if (t == 0) wsum[bn] = t2 / t1;
  float* erow = edge + (size_t)bn * N;
#pragma unroll
  for (int j = 0; j < 4; ++j) erow[t + j * 256] += en[j];
}

// node_out = a @ v (+ wsum broadcast), written bf16 in [B,N,C]
__global__ void __launch_bounds__(32, 2)
k_gemm_av(const bf16* __restrict__ a_bf, const bf16* __restrict__ vT_bf,
          const float* __restrict__ wsum, bf16* __restrict__ nout_bf) {
  int tm = blockIdx.x * 16;
  int bh = blockIdx.y, b = bh / NH, h = bh % NH;
  v8f acc[4]; zero_acc(acc);
  gemm_strip(a_bf + ((size_t)bh * N + tm) * N, N,
             vT_bf + (size_t)bh * DHP * N, N, N / 32, acc);
  int lane = threadIdx.x & 31, cn = lane & 15, mh = (lane >> 4) * 8;
#pragma unroll
  for (int t = 0; t < 4; ++t)
#pragma unroll
    for (int r = 0; r < 8; ++r) {
      int row = tm + mh + r, d = t * 16 + cn;
      if (d < DH) {
        float v = acc[t][r] + wsum[b * N + row];
        nout_bf[((size_t)b * N + row) * C + h * DH + d] = (bf16)v;
      }
    }
}

__global__ void __launch_bounds__(32, 2)
k_gemm_proj(const bf16* __restrict__ nout_bf, const bf16* __restrict__ wp,
            const float* __restrict__ bp, float* __restrict__ xt, int l) {
  int tm = blockIdx.x * 16, tn = blockIdx.y * 64, b = blockIdx.z;
  v8f acc[4]; zero_acc(acc);
  gemm_strip(nout_bf + ((size_t)b * N + tm) * C, C,
             wp + ((size_t)l * C + tn) * C, C, C / 32, acc);
  int lane = threadIdx.x & 31, cn = lane & 15, mh = (lane >> 4) * 8;
#pragma unroll
  for (int t = 0; t < 4; ++t)
#pragma unroll
    for (int r = 0; r < 8; ++r) {
      int row = tm + mh + r, col = tn + t * 16 + cn;
      xt[((size_t)b * N + row) * C + col] = acc[t][r] + bp[l * C + col];
    }
}

// ---------------- launch ----------------
extern "C" void kernel_launch(void* const* d_in, const int* in_sizes, int n_in,
                              void* d_out, int out_size, void* d_ws, size_t ws_size,
                              hipStream_t stream) {
  (void)in_sizes; (void)n_in; (void)out_size; (void)ws_size;
  const float* x      = (const float*)d_in[0];
  const float* dap    = (const float*)d_in[1];
  const float* qk_w   = (const float*)d_in[2];
  const float* fcp_w  = (const float*)d_in[3];
  const float* fcp_b  = (const float*)d_in[4];
  const float* ln1_g  = (const float*)d_in[5];
  const float* ln1_b  = (const float*)d_in[6];
  const float* gln_g  = (const float*)d_in[7];
  const float* gln_b  = (const float*)d_in[8];
  const float* qkv_w  = (const float*)d_in[9];
  const float* qkv_b  = (const float*)d_in[10];
  const float* proj_w = (const float*)d_in[11];
  const float* proj_b = (const float*)d_in[12];
  const float* exp_w  = (const float*)d_in[13];
  const float* exp_b  = (const float*)d_in[14];
  const float* red_w  = (const float*)d_in[15];
  const float* red_b  = (const float*)d_in[16];
  float* outp = (float*)d_out;

  char* ws = (char*)d_ws;
  size_t off = 0;
  auto alloc = [&](size_t bytes) -> char* {
    char* p = ws + off;
    off += (bytes + 255) & ~(size_t)255;
    return p;
  };
  float* xt      = (float*)alloc(sizeof(float) * BB * N * C);
  bf16*  xt_bf   = (bf16*) alloc(sizeof(bf16)  * BB * N * C);
  bf16*  qk_bf   = (bf16*) alloc(sizeof(bf16)  * (size_t)BB * N * TWOC);
  float* edge    = (float*)alloc(sizeof(float) * (size_t)BB * N * N);
  float* attn    = (float*)alloc(sizeof(float) * (size_t)BB * NH * N * N);
  bf16*  a_bf    = (bf16*) alloc(sizeof(bf16)  * (size_t)BB * NH * N * N);
  bf16*  y_bf    = (bf16*) alloc(sizeof(bf16)  * BB * N * C);
  bf16*  q_bf    = (bf16*) alloc(sizeof(bf16)  * BB * NH * N * DHP);
  bf16*  k_bf    = (bf16*) alloc(sizeof(bf16)  * BB * NH * N * DHP);
  bf16*  vT_bf   = (bf16*) alloc(sizeof(bf16)  * BB * NH * DHP * N);
  bf16*  nout_bf = (bf16*) alloc(sizeof(bf16)  * BB * N * C);
  float* wsum    = (float*)alloc(sizeof(float) * BB * N);
  float* s       = (float*)alloc(sizeof(float) * BB * C);
  bf16*  wqk_bf  = (bf16*) alloc(sizeof(bf16)  * TWOC * C);
  bf16*  wqkv_bf = (bf16*) alloc(sizeof(bf16)  * L * THREEC * C);
  bf16*  wp_bf   = (bf16*) alloc(sizeof(bf16)  * L * C * C);

  k_cvt_bf16<<<(TWOC * C + 255) / 256, 256, 0, stream>>>(qk_w, wqk_bf, TWOC * C);
  k_cvt_bf16<<<(L * THREEC * C + 255) / 256, 256, 0, stream>>>(qkv_w, wqkv_bf, L * THREEC * C);
  k_cvt_bf16<<<(L * C * C + 255) / 256, 256, 0, stream>>>(proj_w, wp_bf, L * C * C);
  k_transpose_in<<<(BB * N * C + 255) / 256, 256, 0, stream>>>(x, xt, xt_bf);
  k_da_s<<<(BB * C + 255) / 256, 256, 0, stream>>>(dap, fcp_w, fcp_b, s);
  int zn = BB * NH * N * DHP;
  k_zero_bf16<<<(zn + 255) / 256, 256, 0, stream>>>(q_bf, zn);
  k_zero_bf16<<<(zn + 255) / 256, 256, 0, stream>>>(k_bf, zn);
  k_zero_bf16<<<(zn + 255) / 256, 256, 0, stream>>>(vT_bf, zn);

  dim3 wv(32);
  k_gemm_qk<<<dim3(N / 16, TWOC / 64, BB), wv, 0, stream>>>(xt_bf, wqk_bf, qk_bf);
  k_gemm_edge<<<dim3(N / 16, N / 64, BB), wv, 0, stream>>>(qk_bf, edge);
  k_softmax_rows<<<BB * N, 256, 0, stream>>>(edge);

  for (int l = 0; l < L; ++l) {
    k_norm_node<<<BB * N, C, 0, stream>>>(xt, edge, s, ln1_g, ln1_b, gln_g, gln_b, y_bf, l);
    k_gemm_qkv<<<dim3(N / 16, THREEC / 64, BB), wv, 0, stream>>>(y_bf, wqkv_bf, qkv_b,
                                                                 q_bf, k_bf, vT_bf, l);
    k_gemm_attn<<<dim3(N / 16, N / 64, BB * NH), wv, 0, stream>>>(q_bf, k_bf, edge,
                                                                  exp_w, exp_b, attn, l);
    k_softmax_edge_fused<<<BB * N, 256, 0, stream>>>(attn, a_bf, edge, wsum, red_w, red_b, l);
    k_gemm_av<<<dim3(N / 16, BB * NH), wv, 0, stream>>>(a_bf, vT_bf, wsum, nout_bf);
    k_gemm_proj<<<dim3(N / 16, C / 64, BB), wv, 0, stream>>>(nout_bf, wp_bf, proj_b, xt, l);
  }
  k_final<<<(BB * C * N + 255) / 256, 256, 0, stream>>>(xt, outp);
}
